// FeedForwardSNN_4604204941434
// MI455X (gfx1250) — compile-verified, matrix operand order checked by reading, not run on hardware
//
#include <hip/hip_runtime.h>
#include <stdint.h>

typedef __attribute__((ext_vector_type(16))) _Float16 v16h;
typedef __attribute__((ext_vector_type(8)))  float    v8f;

#define T_STEPS 20
#define DECAY   0.9f
#define THRESH  1.0f
#define H       512        // H1 == H2 == 512 (K for both GEMMs)
#define DOUT    256
#define ROWS    32         // batch rows per block
#define THREADS 128        // 4 waves (wave32)
#define KT_N    16         // 512 / 32 K-steps
#define SSTRIDE 520        // padded f16 row stride for spike arrays (bank-conflict-free)

// K index within a 16x32 f16 A-fragment (CDNA5 ISA 7.12.2):
// lanes 0-15 (g=0): V0..V3 -> K=0..7 pairs, V4..V7 -> K=16..23
// lanes 16-31 (g=1): V0..V3 -> K=8..15,     V4..V7 -> K=24..31
__device__ __forceinline__ int kmap(int d, int g) {
    return (d < 4) ? (2 * d + 8 * g) : (16 + 2 * (d - 4) + 8 * g);
}

union HPack { _Float16 h[2]; uint32_t u; };
union AF    { v16h v; uint32_t u[8]; };
union BF    { v16h v; uint4 q[2]; };

__device__ __forceinline__ v8f vzero() { v8f z = {0,0,0,0,0,0,0,0}; return z; }

__device__ __forceinline__ v8f wmma16(v16h a, v16h b, v8f c) {
    // (neg_a, A, neg_b, B, c_mod, C, reuse_a, reuse_b)
    return __builtin_amdgcn_wmma_f32_16x16x32_f16(false, a, false, b, (short)0, c, false, false);
}

// -------- Prep: split W into f16 hi/lo and pack in B-fragment lane order --------
// Output layout: dword index = ((nt*16 + kt)*32 + lane)*8 + d
// element: n = nt*16 + (lane&15), k = kt*32 + kmap(d, lane>>4) (+1 for high half)
__global__ void snn_pack_w(const float* __restrict__ W, int N, int total,
                           uint32_t* __restrict__ hi, uint32_t* __restrict__ lo) {
    int t = blockIdx.x * blockDim.x + threadIdx.x;
    if (t >= total) return;
    int lane = t & 31;
    int g    = lane >> 4;
    int n    = ((t >> 9) << 4) + (lane & 15);
    int ktb  = ((t >> 5) & 15) * 32;
    uint32_t* ph = hi + (size_t)t * 8;
    uint32_t* pl = lo + (size_t)t * 8;
#pragma unroll
    for (int d = 0; d < 8; ++d) {
        int k = ktb + kmap(d, g);
        float w0 = W[(size_t)k * N + n];
        float w1 = W[(size_t)(k + 1) * N + n];
        _Float16 h0 = (_Float16)w0, h1 = (_Float16)w1;
        _Float16 l0 = (_Float16)(w0 - (float)h0);
        _Float16 l1 = (_Float16)(w1 - (float)h1);
        HPack a; a.h[0] = h0; a.h[1] = h1; ph[d] = a.u;
        HPack b; b.h[0] = l0; b.h[1] = l1; pl[d] = b.u;
    }
}

// -------- Main persistent SNN kernel: 32 rows/block, all 20 steps fused --------
__global__ __launch_bounds__(THREADS, 1)
void snn_main(const float* __restrict__ inp, const float* __restrict__ w_in,
              const uint32_t* __restrict__ w0hi, const uint32_t* __restrict__ w0lo,
              const uint32_t* __restrict__ w1hi, const uint32_t* __restrict__ w1lo,
              float* __restrict__ out) {
    extern __shared__ char smem[];
    float*    lds_m0  = (float*)smem;                            // 32*512 f32  = 65536 B
    _Float16* lds_s0  = (_Float16*)(smem + 65536);               // 32*520 f16  = 33280 B
    _Float16* lds_s1  = (_Float16*)(smem + 65536 + 33280);       // 33280 B
    float*    lds_win = (float*)(smem + 65536 + 66560);          // 5*512 f32   = 10240 B
    float*    lds_in  = (float*)(smem + 65536 + 66560 + 10240);  // 32*4 f32    = 512 B
    const uint32_t* s0u = (const uint32_t*)lds_s0;
    const uint32_t* s1u = (const uint32_t*)lds_s1;

    const int tid  = threadIdx.x;
    const int lane = tid & 31;
    const int wv   = tid >> 5;
    const int lr   = lane & 15;
    const int g    = lane >> 4;
    const int row0 = blockIdx.x * ROWS;

    // ---- one-time init ----
    for (int i = tid; i < 5 * H; i += THREADS) lds_win[i] = w_in[i];
    if (tid < ROWS * 4) lds_in[tid] = inp[row0 * 4 + tid];
    for (int i = tid; i < ROWS * H; i += THREADS) lds_m0[i] = 0.0f;

    v8f m1r[16];   // layer-1 membrane, C-fragment layout, register-resident
    v8f outr[8];   // output integrator, C-fragment layout, register-resident
#pragma unroll
    for (int i = 0; i < 16; ++i) m1r[i] = vzero();
#pragma unroll
    for (int i = 0; i < 8; ++i) outr[i] = vzero();

    __syncthreads();

#pragma unroll 1
    for (int t = 0; t < T_STEPS; ++t) {
        // ---- layer-0 LIF (elementwise in LDS; x recomputed from W_in, 5 FMAs) ----
        for (int i = tid; i < ROWS * H; i += THREADS) {
            int r = i >> 9, c = i & (H - 1);
            const float* ir = lds_in + r * 4;
            float x = ir[0] * lds_win[c] + ir[1] * lds_win[H + c] +
                      ir[2] * lds_win[2 * H + c] + ir[3] * lds_win[3 * H + c] +
                      lds_win[4 * H + c];
            float m = DECAY * lds_m0[i] + x;
            float s = (m > THRESH) ? 1.0f : 0.0f;
            lds_m0[i] = m - s;
            lds_s0[r * SSTRIDE + c] = (_Float16)s;
        }
        __syncthreads();

        // ---- GEMM1: x1 = s0 @ (W0_hi + W0_lo), fused layer-1 LIF epilogue ----
#pragma unroll
        for (int mt = 0; mt < 2; ++mt) {
#pragma unroll
            for (int ng = 0; ng < 2; ++ng) {
                v8f acc[4];
#pragma unroll
                for (int j = 0; j < 4; ++j) acc[j] = vzero();
                const int arow = (mt * 16 + lr) * (SSTRIDE / 2);
                for (int kt = 0; kt < KT_N; ++kt) {
                    AF a;
#pragma unroll
                    for (int d = 0; d < 8; ++d)
                        a.u[d] = s0u[arow + ((kt * 32 + kmap(d, g)) >> 1)];
#pragma unroll
                    for (int j = 0; j < 4; ++j) {
                        int nt = wv * 8 + ng * 4 + j;
                        size_t fb = (size_t)((nt * 16 + kt) * 32 + lane) * 8;
                        const uint4* ph = (const uint4*)(w0hi + fb);
                        const uint4* pl = (const uint4*)(w0lo + fb);
                        BF bl; bl.q[0] = pl[0]; bl.q[1] = pl[1];
                        BF bh; bh.q[0] = ph[0]; bh.q[1] = ph[1];
                        acc[j] = wmma16(a.v, bl.v, acc[j]);
                        acc[j] = wmma16(a.v, bh.v, acc[j]);
                    }
                }
#pragma unroll
                for (int j = 0; j < 4; ++j) {
                    int tix = (mt * 2 + ng) * 4 + j;
                    v8f m = m1r[tix];
                    int colb = (wv * 8 + ng * 4 + j) * 16 + lr;
#pragma unroll
                    for (int r = 0; r < 8; ++r) {
                        float mm = DECAY * m[r] + acc[j][r];
                        float s = (mm > THRESH) ? 1.0f : 0.0f;
                        m[r] = mm - s;
                        int rowi = mt * 16 + g * 8 + r;
                        lds_s1[rowi * SSTRIDE + colb] = (_Float16)s;
                    }
                    m1r[tix] = m;
                }
            }
        }
        __syncthreads();

        // ---- GEMM2: x2 = s1 @ (W1_hi + W1_lo), fused output integrator ----
#pragma unroll
        for (int mt = 0; mt < 2; ++mt) {
            v8f acc[4];
#pragma unroll
            for (int j = 0; j < 4; ++j) acc[j] = vzero();
            const int arow = (mt * 16 + lr) * (SSTRIDE / 2);
            for (int kt = 0; kt < KT_N; ++kt) {
                AF a;
#pragma unroll
                for (int d = 0; d < 8; ++d)
                    a.u[d] = s1u[arow + ((kt * 32 + kmap(d, g)) >> 1)];
#pragma unroll
                for (int j = 0; j < 4; ++j) {
                    int nt = wv * 4 + j;
                    size_t fb = (size_t)((nt * 16 + kt) * 32 + lane) * 8;
                    const uint4* ph = (const uint4*)(w1hi + fb);
                    const uint4* pl = (const uint4*)(w1lo + fb);
                    BF bl; bl.q[0] = pl[0]; bl.q[1] = pl[1];
                    BF bh; bh.q[0] = ph[0]; bh.q[1] = ph[1];
                    acc[j] = wmma16(a.v, bl.v, acc[j]);
                    acc[j] = wmma16(a.v, bh.v, acc[j]);
                }
            }
#pragma unroll
            for (int j = 0; j < 4; ++j) {
                v8f o = outr[mt * 4 + j];
#pragma unroll
                for (int r = 0; r < 8; ++r)
                    o[r] = DECAY * o[r] + acc[j][r];
                outr[mt * 4 + j] = o;
            }
        }
        __syncthreads();
    }

    // ---- writeback out_mem [32, 256] from C-fragment registers ----
#pragma unroll
    for (int mt = 0; mt < 2; ++mt) {
#pragma unroll
        for (int j = 0; j < 4; ++j) {
            int col = (wv * 4 + j) * 16 + lr;
#pragma unroll
            for (int r = 0; r < 8; ++r) {
                int rowi = mt * 16 + g * 8 + r;
                out[(size_t)(row0 + rowi) * DOUT + col] = outr[mt * 4 + j][r];
            }
        }
    }
}

extern "C" void kernel_launch(void* const* d_in, const int* in_sizes, int n_in,
                              void* d_out, int out_size, void* d_ws, size_t ws_size,
                              hipStream_t stream) {
    (void)in_sizes; (void)n_in; (void)out_size; (void)ws_size;
    const float* inp  = (const float*)d_in[0];   // [8192, 4]
    const float* w_in = (const float*)d_in[1];   // [5, 512]
    const float* w0   = (const float*)d_in[2];   // [512, 512]
    const float* w1   = (const float*)d_in[3];   // [512, 256]

    char* ws = (char*)d_ws;
    uint32_t* w0hi = (uint32_t*)(ws);
    uint32_t* w0lo = (uint32_t*)(ws + 512 * 512 * 2);
    uint32_t* w1hi = (uint32_t*)(ws + 512 * 512 * 4);
    uint32_t* w1lo = (uint32_t*)(ws + 512 * 512 * 4 + 512 * 256 * 2);

    const int tot0 = 32 * 16 * 32;  // W0 fragment-lanes (nt=32, kt=16, lane=32)
    const int tot1 = 16 * 16 * 32;  // W1 fragment-lanes
    snn_pack_w<<<(tot0 + 255) / 256, 256, 0, stream>>>(w0, 512, tot0, w0hi, w0lo);
    snn_pack_w<<<(tot1 + 255) / 256, 256, 0, stream>>>(w1, 256, tot1, w1hi, w1lo);

    const size_t smem = 65536 + 33280 + 33280 + 10240 + 512;  // 142848 B (<320KB WGP LDS)
    (void)hipFuncSetAttribute((const void*)snn_main,
                              hipFuncAttributeMaxDynamicSharedMemorySize, (int)smem);
    snn_main<<<8192 / ROWS, THREADS, smem, stream>>>(inp, w_in, w0hi, w0lo, w1hi, w1lo,
                                                     (float*)d_out);
}
